// CNNRNN_62835371540532
// MI455X (gfx1250) — compile-verified, matrix operand order checked by reading, not run on hardware
//
#include <hip/hip_runtime.h>

// ---------------------------------------------------------------------------
// CDNA5 (gfx1250) fused CNN -> MLP -> RNN head using v_wmma_f32_16x16x32_bf16.
// wave32. A-fragment 16-bit layout (ISA 7.12.2):
//   lane L (row M = L%16), element e in v16bf:  K = (e%8) + 16*(e>=8) + 8*(L>=16)
// B-fragment mirrors it with N = L%16. C/D f32: vgpr v -> M = v + 8*(L>=16),
// lane -> N = L%16.
//
// Round-3 change vs round-2: __launch_bounds__(256, 4) on the fused kernel to
// cap the allocator at 256 VGPRs/wave. Round-2 asm showed accumulators placed
// at logical v[400..463] with s_set_vgpr_msb toggling around every WMMA
// (~464 VGPRs/wave) — killing occupancy and polluting the inner loop.
// ---------------------------------------------------------------------------

typedef __attribute__((ext_vector_type(16))) __bf16 v16bf;
typedef __attribute__((ext_vector_type(8)))  float  v8f;

union AFrag {
    v16bf v;
    uint4 q[2];
};

__device__ __forceinline__ unsigned short f32_to_bf16(float f) {
    unsigned int u = __float_as_uint(f);
    u = (u + 0x7FFFu + ((u >> 16) & 1u)) >> 16;   // round-to-nearest-even
    return (unsigned short)u;
}

// ---------------------------------------------------------------------------
// Kernel 0: swizzle an f32 weight matrix into bf16 WMMA-B-fragment layout:
//   dst[((kb*NT + nt)*32 + lane)*16 + e] = B[kb*32 + f(e,lane)][nt*16 + lane%16]
// transposed==1:  B[k][n] = src[n*ld + k]   (for W_in.T / W_out.T)
// transposed==0:  B[k][n] = src[k*ld + n]   (for W, r @ W)
// ---------------------------------------------------------------------------
__global__ void swizzle_b_kernel(const float* __restrict__ src,
                                 unsigned short* __restrict__ dst,
                                 int NT, int ld, int nvalid, int transposed,
                                 int total) {
    int t = blockIdx.x * 256 + threadIdx.x;
    if (t >= total) return;
    int e    = t & 15;
    int lane = (t >> 4) & 31;
    int rec  = t >> 9;
    int nt   = rec % NT;
    int kb   = rec / NT;
    int n = nt * 16 + (lane & 15);
    int k = kb * 32 + (e & 7) + ((e >> 3) << 4) + ((lane >> 4) << 3);
    float v = 0.0f;
    if (n < nvalid)
        v = transposed ? src[(size_t)n * ld + k] : src[(size_t)k * ld + n];
    dst[t] = f32_to_bf16(v);
}

// ---------------------------------------------------------------------------
// Kernel 1: conv(1->16, 3x3, pad=1) + ReLU + maxpool(2x2) -> bf16 [B, 3136]
// One thread per pooled output element (exactly B*16*14*14 threads).
// ---------------------------------------------------------------------------
__global__ __launch_bounds__(256) void conv_pool_kernel(
    const float* __restrict__ x, const float* __restrict__ cw,
    const float* __restrict__ cb, unsigned short* __restrict__ out) {
    int t = blockIdx.x * 256 + threadIdx.x;      // 0 .. 8192*3136-1 (exact)
    int pw = t % 14;
    int tmp = t / 14;
    int ph = tmp % 14; tmp /= 14;
    int c  = tmp & 15;
    int b  = tmp >> 4;

    const float* xb = x + (size_t)b * 784;
    float w[9];
    #pragma unroll
    for (int i = 0; i < 9; ++i) w[i] = cw[c * 9 + i];
    float bias = cb[c];

    float best = 0.0f;                            // ReLU floor
    #pragma unroll
    for (int i = 0; i < 2; ++i) {
        #pragma unroll
        for (int j = 0; j < 2; ++j) {
            int y = 2 * ph + i, xx = 2 * pw + j;
            float a = bias;
            #pragma unroll
            for (int dy = 0; dy < 3; ++dy) {
                int yy = y + dy - 1;
                if (yy < 0 || yy >= 28) continue;
                #pragma unroll
                for (int dx = 0; dx < 3; ++dx) {
                    int xc = xx + dx - 1;
                    if (xc < 0 || xc >= 28) continue;
                    a = fmaf(xb[yy * 28 + xc], w[dy * 3 + dx], a);
                }
            }
            best = fmaxf(best, a);
        }
    }
    out[(size_t)b * 3136 + c * 196 + ph * 14 + pw] = f32_to_bf16(best);
}

// ---------------------------------------------------------------------------
// Kernel 2: fused  E0-GEMM (K=3136) + ReLU + 9x RNN step + output GEMM.
// 128 blocks x 256 threads (8 waves). Block owns 64 batch rows; wave w owns
// row-group (w&3) (16 rows) and N-half (w>>2) (8 N-tiles = 128 cols).
// r lives in 64 f32 acc VGPRs per wave; re-shaped via padded LDS each step.
// ---------------------------------------------------------------------------
#define SR 264   // padded r-stage row stride (halfwords): 528B = 33*16B
#define SA 40    // phase-1 A-stage row stride (halfwords): 80B (bank-safe)

__device__ __forceinline__ void stage_r(unsigned short* lds, const v8f* acc,
                                        int rowgrp, int nt_base, int lo, int hi) {
    #pragma unroll
    for (int nt = 0; nt < 8; ++nt) {
        int n = (nt_base + nt) * 16 + lo;
        #pragma unroll
        for (int v = 0; v < 8; ++v) {
            int m = rowgrp * 16 + v + 8 * hi;
            lds[m * SR + n] = f32_to_bf16(acc[nt][v]);
        }
    }
}

__device__ __forceinline__ void gather_af(const unsigned short* lds,
                                          AFrag* af, int row_local, int hi) {
    #pragma unroll
    for (int kb = 0; kb < 8; ++kb) {
        const unsigned short* p = lds + row_local * SR + kb * 32 + hi * 8;
        af[kb].q[0] = *(const uint4*)p;
        af[kb].q[1] = *(const uint4*)(p + 16);
    }
}

__global__ __launch_bounds__(256, 4) void fused_mlp_rnn_kernel(
    const unsigned short* __restrict__ convbf,   // [8192][3136] bf16
    const unsigned short* __restrict__ bswin,    // [98][16][32][16] bf16
    const unsigned short* __restrict__ bsww,     // [8][16][32][16]  bf16
    const unsigned short* __restrict__ bswout,   // [8][1][32][16]   bf16
    const float* __restrict__ b_in,
    const float* __restrict__ b_out,
    float* __restrict__ outp) {
    // max(2 * 64*SA, 64*SR) halfwords = 16896 hw = 33,792 B
    __shared__ unsigned short lds[64 * SR];

    const int wave = threadIdx.x >> 5;
    const int lane = threadIdx.x & 31;
    const int lo = lane & 15, hi = lane >> 4;
    const int rowgrp  = wave & 3;                 // which 16-row group
    const int nt_base = (wave >> 2) * 8;          // which 8 N-tiles
    const int row_local = rowgrp * 16 + lo;       // this lane's A-frag row
    const int b0 = blockIdx.x * 64;

    float binv[8];
    #pragma unroll
    for (int nt = 0; nt < 8; ++nt) binv[nt] = b_in[(nt_base + nt) * 16 + lo];

    v8f acc[8];
    #pragma unroll
    for (int nt = 0; nt < 8; ++nt) {
        #pragma unroll
        for (int v = 0; v < 8; ++v) acc[nt][v] = binv[nt];   // C0 = b_in
    }

    // ---------------- Phase 1: E0 = conv_flat @ W_in.T + b_in --------------
    // Double-buffered LDS A-stage: 64 rows x 32 K bf16 (64B/row) per buffer.
    // 256 threads each move one 16B chunk; prefetch of tile kb+1 overlaps the
    // WMMA chain of tile kb. One barrier per iteration.
    {
        const int srow  = threadIdx.x >> 2;       // 0..63
        const int squad = threadIdx.x & 3;        // 16B chunk within row
        const unsigned short* abase =
            convbf + (size_t)(b0 + srow) * 3136 + squad * 8;
        unsigned short* buf0 = lds;               // 64*SA hw = 5120B
        unsigned short* buf1 = lds + 64 * SA;

        uint4 areg = *(const uint4*)abase;        // prefetch tile kb = 0
        for (int kb = 0; kb < 98; ++kb) {
            unsigned short* sbuf = (kb & 1) ? buf1 : buf0;
            *(uint4*)(sbuf + srow * SA + squad * 8) = areg;
            __syncthreads();
            if (kb + 1 < 98)                      // prefetch next tile
                areg = *(const uint4*)(abase + (size_t)(kb + 1) * 32);

            AFrag a;
            {
                const unsigned short* p = sbuf + row_local * SA + hi * 8;
                a.q[0] = *(const uint4*)p;
                a.q[1] = *(const uint4*)(p + 16);
            }
            #pragma unroll
            for (int nt = 0; nt < 8; ++nt) {
                AFrag bfr;
                const uint4* bp = (const uint4*)(
                    bswin + (((size_t)(kb * 16 + nt_base + nt) * 32 + lane) << 4));
                bfr.q[0] = bp[0];
                bfr.q[1] = bp[1];
                acc[nt] = __builtin_amdgcn_wmma_f32_16x16x32_bf16(
                    false, a.v, false, bfr.v, (short)0, acc[nt], false, false);
            }
        }
    }
    #pragma unroll
    for (int nt = 0; nt < 8; ++nt) {
        #pragma unroll
        for (int v = 0; v < 8; ++v) acc[nt][v] = fmaxf(acc[nt][v], 0.0f);
    }

    // ---------------- Phase 2: 9x  r = relu(r @ W + b_in + r) --------------
    for (int step = 0; step < 9; ++step) {
        __syncthreads();                          // prior readers done
        stage_r(lds, acc, rowgrp, nt_base, lo, hi);
        __syncthreads();
        AFrag af[8];
        gather_af(lds, af, row_local, hi);
        #pragma unroll
        for (int nt = 0; nt < 8; ++nt) {
            v8f c;
            #pragma unroll
            for (int v = 0; v < 8; ++v) c[v] = acc[nt][v] + binv[nt]; // r + b_in
            #pragma unroll
            for (int kb = 0; kb < 8; ++kb) {
                AFrag bfr;
                const uint4* bp = (const uint4*)(
                    bsww + (((size_t)(kb * 16 + nt_base + nt) * 32 + lane) << 4));
                bfr.q[0] = bp[0];
                bfr.q[1] = bp[1];
                c = __builtin_amdgcn_wmma_f32_16x16x32_bf16(
                    false, af[kb].v, false, bfr.v, (short)0, c, false, false);
            }
            #pragma unroll
            for (int v = 0; v < 8; ++v) acc[nt][v] = fmaxf(c[v], 0.0f);
        }
    }

    // ---------------- Phase 3: out = r @ W_out.T + b_out (N padded to 16) --
    __syncthreads();
    stage_r(lds, acc, rowgrp, nt_base, lo, hi);
    __syncthreads();
    if (nt_base == 0) {                           // waves 0..3 (wave-uniform)
        AFrag af[8];
        gather_af(lds, af, row_local, hi);
        v8f c;
        #pragma unroll
        for (int v = 0; v < 8; ++v) c[v] = 0.0f;
        #pragma unroll
        for (int kb = 0; kb < 8; ++kb) {
            AFrag bfr;
            const uint4* bp =
                (const uint4*)(bswout + (((size_t)kb * 32 + lane) << 4));
            bfr.q[0] = bp[0];
            bfr.q[1] = bp[1];
            c = __builtin_amdgcn_wmma_f32_16x16x32_bf16(
                false, af[kb].v, false, bfr.v, (short)0, c, false, false);
        }
        if (lo < 10) {
            float bo = b_out[lo];
            #pragma unroll
            for (int v = 0; v < 8; ++v) {
                int m = rowgrp * 16 + v + 8 * hi;
                outp[(size_t)(b0 + m) * 10 + lo] = c[v] + bo;
            }
        }
    }
}

// ---------------------------------------------------------------------------
// Host launcher. Inputs (setup_inputs order):
//   0:x 1:conv_w 2:conv_b 3:W_in 4:b_in 5:W_out 6:b_out 7:W
// Workspace: bf16 conv_flat (51.4MB) + swizzled bf16 weights (~1.75MB).
// ---------------------------------------------------------------------------
extern "C" void kernel_launch(void* const* d_in, const int* in_sizes, int n_in,
                              void* d_out, int out_size, void* d_ws,
                              size_t ws_size, hipStream_t stream) {
    const float* x      = (const float*)d_in[0];
    const float* conv_w = (const float*)d_in[1];
    const float* conv_b = (const float*)d_in[2];
    const float* W_in   = (const float*)d_in[3];
    const float* b_in   = (const float*)d_in[4];
    const float* W_out  = (const float*)d_in[5];
    const float* b_out  = (const float*)d_in[6];
    const float* W      = (const float*)d_in[7];
    float* outp = (float*)d_out;

    unsigned char* ws = (unsigned char*)d_ws;
    const size_t CONV_BYTES  = (size_t)8192 * 3136 * 2;       // 51,380,224
    const size_t BSWIN_BYTES = (size_t)98 * 16 * 512 * 2;     //  1,605,632
    const size_t BSWW_BYTES  = (size_t)8 * 16 * 512 * 2;      //    131,072
    unsigned short* convbf = (unsigned short*)ws;
    unsigned short* bswin  = (unsigned short*)(ws + CONV_BYTES);
    unsigned short* bsww   = (unsigned short*)(ws + CONV_BYTES + BSWIN_BYTES);
    unsigned short* bswout = (unsigned short*)(ws + CONV_BYTES + BSWIN_BYTES + BSWW_BYTES);

    // weight swizzles (tiny)
    swizzle_b_kernel<<<3136, 256, 0, stream>>>(W_in,  bswin, 16, 3136, 256, 1, 802816);
    swizzle_b_kernel<<<256,  256, 0, stream>>>(W,     bsww,  16, 256,  256, 0, 65536);
    swizzle_b_kernel<<<16,   256, 0, stream>>>(W_out, bswout, 1, 256,  10,  1, 4096);

    // conv + relu + pool -> bf16 conv_flat  (8192*3136 = 25,690,112 threads)
    conv_pool_kernel<<<100352, 256, 0, stream>>>(x, conv_w, conv_b, convbf);

    // fused GEMM + RNN + head: 128 blocks * 8 waves, 64 batch rows per block
    fused_mlp_rnn_kernel<<<128, 256, 0, stream>>>(convbf, bswin, bsww, bswout,
                                                  b_in, b_out, outp);
}
